// DecoderLM_81844896792978
// MI455X (gfx1250) — matrix operand order repaired and checked by reading, not compiled, not run on hardware
//
#include <hip/hip_runtime.h>
#include <hip/hip_bf16.h>
#include <math.h>

// ---------------- model constants (match reference) ----------------
#define MV 32000
#define MD 512
#define MH 8
#define MF 2048
#define ML 4
#define MB 4
#define MT 1024
#define MDK 64
#define MM (MB * MT)   // 4096 token rows

typedef __attribute__((ext_vector_type(16))) _Float16 v16h;
typedef __attribute__((ext_vector_type(8)))  _Float16 v8h;
typedef __attribute__((ext_vector_type(2)))  _Float16 v2h;
typedef __attribute__((ext_vector_type(8)))  float    v8f;

#define LSTRIDE 40   // LDS row stride in halfs: 80B = 5*16B -> all chunks 16B aligned

__device__ __forceinline__ v8f wmma_f16(v16h a, v16h b, v8f c) {
    return __builtin_amdgcn_wmma_f32_16x16x32_f16(
        false, a, false, b, (short)0, c, false, false);
}

// A-style fragment (also P): chunks at row[half*8..+7] and row[16+half*8..+7]
__device__ __forceinline__ v16h frag_a(const _Float16* row, int half) {
    v8h lo = *(const v8h*)(row + half * 8);
    v8h hi = *(const v8h*)(row + 16 + half * 8);
    return __builtin_shufflevector(lo, hi, 0,1,2,3,4,5,6,7,8,9,10,11,12,13,14,15);
}

// B-style fragment: chunks at row[half*8..+7] and row[16+half*8..+7]
__device__ __forceinline__ v16h frag_b(const _Float16* row, int half) {
    v8h lo = *(const v8h*)(row + half * 8);
    v8h hi = *(const v8h*)(row + 16 + half * 8);
    return __builtin_shufflevector(lo, hi, 0,1,2,3,4,5,6,7,8,9,10,11,12,13,14,15);
}

__device__ __forceinline__ float gelu_exact(float x) {
    return 0.5f * x * (1.0f + erff(x * 0.70710678118654752f));
}

// ---------------- embedding + sinusoidal positional encoding ----------------
__global__ __launch_bounds__(256) void embed_pe_kernel(const int* __restrict__ x,
                                                       const float* __restrict__ embed,
                                                       float* __restrict__ h) {
    size_t i = (size_t)blockIdx.x * 256 + threadIdx.x;   // over B*T*D
    int d  = (int)(i % MD);
    int bt = (int)(i / MD);
    int tt = bt % MT;
    int tok = x[bt];
    float e = (tok == 0) ? 0.0f : embed[(size_t)tok * MD + d];
    int i2 = d >> 1;
    float ang = (float)tt * __expf(-(float)(2 * i2) * (9.210340371976184f / (float)MD));
    float pe = (d & 1) ? __cosf(ang) : __sinf(ang);
    h[i] = e + pe;
}

// ---------------- generic WMMA GEMM:  C[M,N] = act(A[M,K] @ W[K,N] + bias) ----------------
// grid = (N/64, M/64), block = 256 (8 waves); each wave owns two 16x16 accumulators.
template <bool GELU>
__global__ __launch_bounds__(256) void wmma_gemm_kernel(const float* __restrict__ A,
                                                        const float* __restrict__ W,
                                                        const float* __restrict__ bias,
                                                        float* __restrict__ C,
                                                        int M, int N, int K) {
    __shared__ _Float16 As[64][LSTRIDE];   // [m][k]
    __shared__ _Float16 Bs[64][LSTRIDE];   // [n][k]  (transposed)

    const int t    = threadIdx.x;
    const int wave = t >> 5;
    const int lane = t & 31;
    const int n16  = lane & 15;
    const int half = lane >> 4;
    const int wr   = wave & 3;
    const int wc   = wave >> 2;
    const int bm   = blockIdx.y * 64;
    const int bn   = blockIdx.x * 64;

    v8f c0 = {}; v8f c1 = {};

    for (int k0 = 0; k0 < K; k0 += 32) {
        // --- stage A tile 64x32 as packed f16 pairs (4 pairs / thread) ---
        for (int i = t; i < 64 * 16; i += 256) {
            int r = i >> 4, col2 = (i & 15) * 2;
            const float* src = A + (size_t)(bm + r) * K + (k0 + col2);
            v2h p; p[0] = (_Float16)src[0]; p[1] = (_Float16)src[1];
            *(v2h*)&As[r][col2] = p;
        }
        // --- stage B tile 32x64 transposed, packed along k ---
        for (int i = t; i < 64 * 16; i += 256) {
            int n = i & 63, k2 = (i >> 6) * 2;
            const float* src = W + (size_t)(k0 + k2) * N + (bn + n);
            v2h p; p[0] = (_Float16)src[0]; p[1] = (_Float16)src[N];
            *(v2h*)&Bs[n][k2] = p;
        }
        if (k0 + 32 < K) {
            int r = t >> 2;
            __builtin_prefetch(A + (size_t)(bm + r) * K + (k0 + 32), 0, 3);
            __builtin_prefetch(W + (size_t)(k0 + 32 + (t & 31)) * N + bn, 0, 3);
        }
        __syncthreads();

        v16h a  = frag_a(&As[wr * 16 + n16][0], half);
        v16h b0 = frag_b(&Bs[wc * 32 + n16][0], half);
        v16h b1 = frag_b(&Bs[wc * 32 + 16 + n16][0], half);
        c0 = wmma_f16(a, b0, c0);
        c1 = wmma_f16(a, b1, c1);
        __syncthreads();
    }

    // --- epilogue: C layout m = r + 8*half, n = lane%16 ---
    const int gm0 = bm + wr * 16;
    const int gn0 = bn + wc * 32;
    const float bias0 = bias ? bias[gn0 + n16] : 0.0f;
    const float bias1 = bias ? bias[gn0 + 16 + n16] : 0.0f;
    #pragma unroll
    for (int r = 0; r < 8; ++r) {
        int gm = gm0 + r + 8 * half;
        float x0 = c0[r] + bias0;
        float x1 = c1[r] + bias1;
        if (GELU) { x0 = gelu_exact(x0); x1 = gelu_exact(x1); }
        C[(size_t)gm * N + gn0 + n16]      = x0;
        C[(size_t)gm * N + gn0 + 16 + n16] = x1;
    }
}

// ---------------- WMMA flash attention ----------------
// grid = (T/16, H, B), block = 32 (one wave per 16-query tile).
__global__ __launch_bounds__(32) void attention_kernel(const float* __restrict__ q,
                                                       const float* __restrict__ k,
                                                       const float* __restrict__ v,
                                                       float* __restrict__ out) {
    const int b    = blockIdx.z;
    const int hh   = blockIdx.y;
    const int qt0  = blockIdx.x * 16;
    const int lane = threadIdx.x;
    const int n16  = lane & 15;
    const int half = lane >> 4;

    __shared__ _Float16 vt[64][LSTRIDE];     // V tile transposed: [dk][key]
    __shared__ _Float16 plds[16][LSTRIDE];   // P tile: [m][key]

    // Q fragments (two dk-chunks of 32), softmax scale 1/8 folded in
    v16h aQ0, aQ1;
    {
        size_t qrow = ((size_t)(b * MT + qt0 + n16)) * MD + hh * MDK;
        #pragma unroll
        for (int e = 0; e < 16; ++e) {
            int kk = half * 16 + e;
            aQ0[e] = (_Float16)(q[qrow + kk] * 0.125f);
            aQ1[e] = (_Float16)(q[qrow + 32 + kk] * 0.125f);
        }
    }

    float mi[8], li[8];
    v8f o0 = {}, o1 = {}, o2 = {}, o3 = {};
    #pragma unroll
    for (int r = 0; r < 8; ++r) { mi[r] = -1e30f; li[r] = 0.0f; }

    const int nkb = (qt0 + 15) / 32 + 1;
    for (int kb = 0; kb < nkb; ++kb) {
        const int kt0 = kb * 32;

        // stage V tile transposed, packed pairs of keys (32 pairs / lane)
        for (int i = lane; i < 64 * 16; i += 32) {
            int cc = i & 63, rr = (i >> 6) * 2;                   // cc = dk, rr = key pair
            size_t vbase = ((size_t)(b * MT + kt0 + rr)) * MD + hh * MDK + cc;
            v2h p; p[0] = (_Float16)v[vbase]; p[1] = (_Float16)v[vbase + MD];
            *(v2h*)&vt[cc][rr] = p;
        }

        // K^T B-fragments straight from global (16 consecutive dk per lane)
        v16h bK00, bK01, bK10, bK11;
        {
            size_t k0row = ((size_t)(b * MT + kt0 + n16)) * MD + hh * MDK + half * 16;
            size_t k1row = ((size_t)(b * MT + kt0 + 16 + n16)) * MD + hh * MDK + half * 16;
            #pragma unroll
            for (int e = 0; e < 16; ++e) {
                bK00[e] = (_Float16)k[k0row + e];
                bK01[e] = (_Float16)k[k0row + 32 + e];
                bK10[e] = (_Float16)k[k1row + e];
                bK11[e] = (_Float16)k[k1row + 32 + e];
            }
        }

        v8f s0 = {}, s1 = {};
        s0 = wmma_f16(aQ0, bK00, s0);
        s0 = wmma_f16(aQ1, bK01, s0);
        s1 = wmma_f16(aQ0, bK10, s1);
        s1 = wmma_f16(aQ1, bK11, s1);

        // causal mask (uniform branch; only trailing block needs it)
        if (kt0 + 31 > qt0) {
            #pragma unroll
            for (int r = 0; r < 8; ++r) {
                int query = qt0 + r + 8 * half;
                if (kt0 + n16 > query)      s0[r] = -1e30f;
                if (kt0 + 16 + n16 > query) s1[r] = -1e30f;
            }
        }

        // online softmax; row m = r + 8*half lives in the 16 lanes of this half
        #pragma unroll
        for (int r = 0; r < 8; ++r) {
            float mx = fmaxf(s0[r], s1[r]);
            #pragma unroll
            for (int d = 1; d < 16; d <<= 1) mx = fmaxf(mx, __shfl_xor(mx, d, 32));
            float newm = fmaxf(mi[r], mx);
            float p0 = __expf(s0[r] - newm);
            float p1 = __expf(s1[r] - newm);
            float corr = __expf(mi[r] - newm);
            float rs = p0 + p1;
            #pragma unroll
            for (int d = 1; d < 16; d <<= 1) rs += __shfl_xor(rs, d, 32);
            li[r] = li[r] * corr + rs;
            mi[r] = newm;
            s0[r] = p0; s1[r] = p1;
            o0[r] *= corr; o1[r] *= corr; o2[r] *= corr; o3[r] *= corr;
        }

        __syncthreads();
        #pragma unroll
        for (int r = 0; r < 8; ++r) {
            int m = r + 8 * half;
            plds[m][n16]      = (_Float16)s0[r];
            plds[m][16 + n16] = (_Float16)s1[r];
        }
        __syncthreads();

        v16h aP  = frag_a(&plds[n16][0], half);
        v16h bV0 = frag_b(&vt[n16][0], half);
        v16h bV1 = frag_b(&vt[16 + n16][0], half);
        v16h bV2 = frag_b(&vt[32 + n16][0], half);
        v16h bV3 = frag_b(&vt[48 + n16][0], half);
        o0 = wmma_f16(aP, bV0, o0);
        o1 = wmma_f16(aP, bV1, o1);
        o2 = wmma_f16(aP, bV2, o2);
        o3 = wmma_f16(aP, bV3, o3);
        __syncthreads();
    }

    // normalize and scatter output rows
    #pragma unroll
    for (int r = 0; r < 8; ++r) {
        float inv = 1.0f / li[r];
        int m = r + 8 * half;
        size_t orow = ((size_t)(b * MT + qt0 + m)) * MD + hh * MDK;
        out[orow + n16]      = o0[r] * inv;
        out[orow + 16 + n16] = o1[r] * inv;
        out[orow + 32 + n16] = o2[r] * inv;
        out[orow + 48 + n16] = o3[r] * inv;
    }
}

// ---------------- fused residual-add + layernorm (in place on h) ----------------
__global__ __launch_bounds__(256) void layernorm_kernel(float* __restrict__ h,
                                                        const float* __restrict__ add,
                                                        const float* __restrict__ s,
                                                        const float* __restrict__ bi) {
    const int row = blockIdx.x;
    const int t   = threadIdx.x;
    const size_t base = (size_t)row * MD;
    float x0 = h[base + t];
    float x1 = h[base + t + 256];
    if (add) { x0 += add[base + t]; x1 += add[base + t + 256]; }

    __shared__ float red[256];
    red[t] = x0 + x1;
    __syncthreads();
    for (int o = 128; o > 0; o >>= 1) { if (t < o) red[t] += red[t + o]; __syncthreads(); }
    float mean = red[0] * (1.0f / MD);
    __syncthreads();

    float d0 = x0 - mean, d1 = x1 - mean;
    red[t] = d0 * d0 + d1 * d1;
    __syncthreads();
    for (int o = 128; o > 0; o >>= 1) { if (t < o) red[t] += red[t + o]; __syncthreads(); }
    float inv = rsqrtf(red[0] * (1.0f / MD) + 1e-5f);

    h[base + t]       = d0 * inv * s[t] + bi[t];
    h[base + t + 256] = d1 * inv * s[t + 256] + bi[t + 256];
}

// ---------------- driver ----------------
extern "C" void kernel_launch(void* const* d_in, const int* in_sizes, int n_in,
                              void* d_out, int out_size, void* d_ws, size_t ws_size,
                              hipStream_t stream) {
    (void)in_sizes; (void)n_in; (void)out_size; (void)ws_size;

    const int*   x     = (const int*)  d_in[0];
    const float* embed = (const float*)d_in[1];
    const float* Wq    = (const float*)d_in[2];
    const float* bq    = (const float*)d_in[3];
    const float* Wk    = (const float*)d_in[4];
    const float* bk    = (const float*)d_in[5];
    const float* Wv    = (const float*)d_in[6];
    const float* bv    = (const float*)d_in[7];
    const float* Wo    = (const float*)d_in[8];
    const float* bo    = (const float*)d_in[9];
    const float* ln1s  = (const float*)d_in[10];
    const float* ln1b  = (const float*)d_in[11];
    const float* W1    = (const float*)d_in[12];
    const float* b1    = (const float*)d_in[13];
    const float* W2    = (const float*)d_in[14];
    const float* b2    = (const float*)d_in[15];
    const float* ln2s  = (const float*)d_in[16];
    const float* ln2b  = (const float*)d_in[17];
    const float* lnfs  = (const float*)d_in[18];
    const float* lnfb  = (const float*)d_in[19];
    const float* Wl    = (const float*)d_in[20];
    const float* bl    = (const float*)d_in[21];
    float* logits = (float*)d_out;

    const size_t BTD = (size_t)MM * MD;
    float* h  = (float*)d_ws;
    float* qb = h  + BTD;
    float* kb = qb + BTD;
    float* vb = kb + BTD;
    float* ab = vb + BTD;
    float* ob = ab + BTD;
    float* ff = ob + BTD;   // [MM, MF]

    embed_pe_kernel<<<dim3((unsigned)(BTD / 256)), 256, 0, stream>>>(x, embed, h);

    const dim3 gDD(MD / 64, MM / 64);
    const dim3 gDF(MF / 64, MM / 64);
    const dim3 gLM(MV / 64, MM / 64);
    const dim3 gAtt(MT / 16, MH, MB);

    for (int l = 0; l < ML; ++l) {
        const float* wq = Wq + (size_t)l * MD * MD;
        const float* wk = Wk + (size_t)l * MD * MD;
        const float* wv = Wv + (size_t)l * MD * MD;
        const float* wo = Wo + (size_t)l * MD * MD;
        const float* w1 = W1 + (size_t)l * MD * MF;
        const float* w2 = W2 + (size_t)l * MF * MD;

        wmma_gemm_kernel<false><<<gDD, 256, 0, stream>>>(h, wq, bq + l * MD, qb, MM, MD, MD);
        wmma_gemm_kernel<false><<<gDD, 256, 0, stream>>>(h, wk, bk + l * MD, kb, MM, MD, MD);
        wmma_gemm_kernel<false><<<gDD, 256, 0, stream>>>(h, wv, bv + l * MD, vb, MM, MD, MD);

        attention_kernel<<<gAtt, 32, 0, stream>>>(qb, kb, vb, ab);

        wmma_gemm_kernel<false><<<gDD, 256, 0, stream>>>(ab, wo, bo + l * MD, ob, MM, MD, MD);
        layernorm_kernel<<<MM, 256, 0, stream>>>(h, ob, ln1s + l * MD, ln1b + l * MD);

        wmma_gemm_kernel<true ><<<gDF, 256, 0, stream>>>(h, w1, b1 + l * MF, ff, MM, MF, MD);
        wmma_gemm_kernel<false><<<gDD, 256, 0, stream>>>(ff, w2, b2 + l * MD, ob, MM, MD, MF);
        layernorm_kernel<<<MM, 256, 0, stream>>>(h, ob, ln2s + l * MD, ln2b + l * MD);
    }

    layernorm_kernel<<<MM, 256, 0, stream>>>(h, nullptr, lnfs, lnfb);
    wmma_gemm_kernel<false><<<gLM, 256, 0, stream>>>(h, Wl, bl, logits, MM, MV, MD);
}
